// ConMambaCTC_30769145709300
// MI455X (gfx1250) — compile-verified, matrix operand order checked by reading, not run on hardware
//
#include <hip/hip_runtime.h>
#include <hip/hip_bf16.h>

typedef __attribute__((ext_vector_type(16))) _Float16 v16h;
typedef __attribute__((ext_vector_type(8)))  _Float16 v8h;
typedef __attribute__((ext_vector_type(8)))  float    v8f;

// ---------- model constants ----------
#define BD      4
#define TFULL   2048
#define MEL     80
#define DMODEL  512
#define DINNER  1024
#define STATE   16
#define DTRANK  32
#define NBLK    8
#define VOCAB   2048
#define LD      512          // T/4
#define MROWS   (BD*LD)      // 2048

__device__ __forceinline__ float gelu_exact(float x) {
    return 0.5f * x * (1.f + erff(x * 0.70710678118654752f));
}
__device__ __forceinline__ float silu(float x) {
    return x / (1.f + __expf(-x));
}
__device__ __forceinline__ float softplus_f(float x) {
    return (x > 20.f) ? x : log1pf(__expf(x));
}

// =====================================================================
// Batched weight transpose + f32->f16 convert:  W(K,N) -> Wt(N,K)
// 32x32 LDS tile, block (32,8), grid (N/32, K/32, nb). Dims are all
// multiples of 32 in this model.
// =====================================================================
__global__ __launch_bounds__(256) void transpose_cvt_f16(
    const float* __restrict__ W, _Float16* __restrict__ Wt, int K, int N)
{
    __shared__ float t[32][33];
    const size_t bofs = (size_t)blockIdx.z * K * N;
    const float*    Wb  = W  + bofs;
    _Float16*       Wtb = Wt + bofs;
    const int n0 = blockIdx.x * 32, k0 = blockIdx.y * 32;
    const int tx = threadIdx.x, ty = threadIdx.y;
    #pragma unroll
    for (int i = 0; i < 4; ++i)
        t[ty + 8*i][tx] = Wb[(size_t)(k0 + ty + 8*i) * N + n0 + tx];
    __syncthreads();
    #pragma unroll
    for (int i = 0; i < 4; ++i)
        Wtb[(size_t)(n0 + ty + 8*i) * K + k0 + tx] = (_Float16)t[tx][ty + 8*i];
}

// =====================================================================
// WMMA GEMM: C[M,N] = A[M,K](f32) x Bt[N,K](f16, pre-transposed)
// block tile 64x128, BK=32, 256 threads = 8 waves, wave tile 32x32.
// As[m][k] and Bs[n][k] both K-contiguous -> all fragment loads are
// ds_load_b128 pairs; all staging is b128 loads/stores.
// FULL=true: tiles fully in-bounds (every GEMM here except x_proj N=64).
// epilogue: optional bias[N], act (0=none,1=softplus), accumulate (C+=).
// =====================================================================
template<bool FULL>
__global__ __launch_bounds__(256) void gemm_wmma_f16(
    const float* __restrict__ A, int lda,
    const _Float16* __restrict__ Bt, int ldbt,   // ldbt == Kdim
    float* __restrict__ C, int ldc,
    int Mdim, int Ndim, int Kdim,
    const float* __restrict__ bias, int act, int accumulate)
{
    __shared__ _Float16 As[64][40];     // 64 rows  x 32 k (+pad)
    __shared__ _Float16 Bs[128][40];    // 128 cols x 32 k (+pad)

    const int tid    = threadIdx.x;
    const int blockM = blockIdx.y * 64;
    const int blockN = blockIdx.x * 128;

    // loaders
    const int arow  = tid >> 2;         // 0..63
    const int acol0 = (tid & 3) * 8;    // 0,8,16,24
    const int bn    = tid >> 1;         // 0..127  (n index)
    const int bk0   = (tid & 1) * 16;   // 0 | 16  (k index)

    // wave mapping
    const int lane = tid & 31;
    const int wave = tid >> 5;
    const int wm   = (wave >> 2) * 32;  // 0 | 32
    const int wn   = (wave & 3) * 32;   // 0..96
    const int m    = lane & 15;
    const int kb   = (lane >> 4) * 8;   // 0 | 8

    v8f acc[2][2] = {};

    for (int k0 = 0; k0 < Kdim; k0 += 32) {
        // ---- stage A tile (f32 -> f16), K-contiguous ----
        if (FULL) {
            const float* aptr = A + (size_t)(blockM + arow) * lda + k0 + acol0;
            if (k0 + 64 <= Kdim) __builtin_prefetch(aptr + 32, 0, 1);
            const float4 a0 = ((const float4*)aptr)[0];
            const float4 a1 = ((const float4*)aptr)[1];
            v8h hv;
            hv[0] = (_Float16)a0.x; hv[1] = (_Float16)a0.y;
            hv[2] = (_Float16)a0.z; hv[3] = (_Float16)a0.w;
            hv[4] = (_Float16)a1.x; hv[5] = (_Float16)a1.y;
            hv[6] = (_Float16)a1.z; hv[7] = (_Float16)a1.w;
            *(v8h*)&As[arow][acol0] = hv;
        } else {
            const int gr = blockM + arow;
            #pragma unroll
            for (int j = 0; j < 8; ++j) {
                const int gc = k0 + acol0 + j;
                float v = (gr < Mdim && gc < Kdim) ? A[(size_t)gr * lda + gc] : 0.f;
                As[arow][acol0 + j] = (_Float16)v;
            }
        }
        // ---- stage B tile (f16, already [n][k]) : pure b128 traffic ----
        if (FULL) {
            const _Float16* bp = Bt + (size_t)(blockN + bn) * ldbt + k0 + bk0;
            if (k0 + 64 <= Kdim) __builtin_prefetch(bp + 32, 0, 1);
            *(v8h*)&Bs[bn][bk0]     = *(const v8h*)bp;
            *(v8h*)&Bs[bn][bk0 + 8] = *(const v8h*)(bp + 8);
        } else {
            const int gn = blockN + bn;
            #pragma unroll
            for (int j = 0; j < 16; ++j) {
                const int gk = k0 + bk0 + j;
                _Float16 v = (_Float16)0.f;
                if (gn < Ndim && gk < Kdim) v = Bt[(size_t)gn * ldbt + gk];
                Bs[bn][bk0 + j] = v;
            }
        }
        __syncthreads();

        // ---- fragments: half h -> k = (h<8?0:16) + kb + (h&7) ----
        v16h afrag[2], bfrag[2];
        #pragma unroll
        for (int ti = 0; ti < 2; ++ti) {
            const v8h alo = *(const v8h*)&As[wm + ti * 16 + m][kb];
            const v8h ahi = *(const v8h*)&As[wm + ti * 16 + m][16 + kb];
            const v8h blo = *(const v8h*)&Bs[wn + ti * 16 + m][kb];
            const v8h bhi = *(const v8h*)&Bs[wn + ti * 16 + m][16 + kb];
            #pragma unroll
            for (int j = 0; j < 8; ++j) {
                afrag[ti][j] = alo[j]; afrag[ti][8 + j] = ahi[j];
                bfrag[ti][j] = blo[j]; bfrag[ti][8 + j] = bhi[j];
            }
        }
        #pragma unroll
        for (int tm = 0; tm < 2; ++tm)
            #pragma unroll
            for (int tn = 0; tn < 2; ++tn)
                acc[tm][tn] = __builtin_amdgcn_wmma_f32_16x16x32_f16(
                    false, afrag[tm], false, bfrag[tn],
                    (short)0, acc[tm][tn], false, false);
        __syncthreads();
    }

    // epilogue + store: C/D layout: VGPR r -> row kb + r, col = lane&15
    #pragma unroll
    for (int tm = 0; tm < 2; ++tm) {
        #pragma unroll
        for (int tn = 0; tn < 2; ++tn) {
            #pragma unroll
            for (int r = 0; r < 8; ++r) {
                const int row = blockM + wm + tm * 16 + kb + r;
                const int col = blockN + wn + tn * 16 + m;
                if (FULL || (row < Mdim && col < Ndim)) {
                    float v = acc[tm][tn][r];
                    if (bias) v += bias[col];
                    if (act == 1) v = softplus_f(v);
                    const size_t idx = (size_t)row * ldc + col;
                    if (accumulate) v += C[idx];
                    C[idx] = v;
                }
            }
        }
    }
}

// =====================================================================
// Frontend conv1: feats(B,T,80) -> y1(B,512,1024)  stride2 pad1 K3, GELU
// =====================================================================
__global__ __launch_bounds__(256) void frontend_conv1(
    const float* __restrict__ feats, const float* __restrict__ w,
    const float* __restrict__ b, float* __restrict__ y1)
{
    const int idx = blockIdx.x * 256 + threadIdx.x;  // B*512*1024
    const int l  = idx & 1023;
    const int co = (idx >> 10) & 511;
    const int bb = idx >> 19;
    float acc = b[co];
    const int t0 = 2 * l - 1;
    #pragma unroll
    for (int k = 0; k < 3; ++k) {
        const int t = t0 + k;
        if (t >= 0 && t < TFULL) {
            const float* xr = feats + ((size_t)bb * TFULL + t) * MEL;
            const float* wr = w + (size_t)co * MEL * 3 + k;
            for (int ci = 0; ci < MEL; ++ci) acc += wr[ci * 3] * xr[ci];
        }
    }
    y1[idx] = gelu_exact(acc);
}

// =====================================================================
// Frontend conv2: y1(B,512,1024) -> x(B*L, 512)   stride2 pad1 K3, GELU
// =====================================================================
__global__ __launch_bounds__(256) void frontend_conv2(
    const float* __restrict__ y1, const float* __restrict__ w,
    const float* __restrict__ b, float* __restrict__ x)
{
    const int idx = blockIdx.x * 256 + threadIdx.x;  // B*512*512
    const int l  = idx & 511;
    const int co = (idx >> 9) & 511;
    const int bb = idx >> 18;
    float acc = b[co];
    const int t0 = 2 * l - 1;
    #pragma unroll
    for (int k = 0; k < 3; ++k) {
        const int t = t0 + k;
        if (t >= 0 && t < 1024) {
            const float* xr = y1 + (size_t)bb * 512 * 1024 + t;
            const float* wr = w + (size_t)co * 512 * 3 + k;
            for (int ci = 0; ci < 512; ++ci) acc += wr[ci * 3] * xr[(size_t)ci * 1024];
        }
    }
    x[((size_t)(bb * LD + l)) * DMODEL + co] = gelu_exact(acc);
}

// =====================================================================
// RMSNorm over last dim (D=512), one workgroup per row
// =====================================================================
__global__ __launch_bounds__(256) void rmsnorm_k(
    const float* __restrict__ x, const float* __restrict__ nw,
    float* __restrict__ xn)
{
    const int row = blockIdx.x;
    __shared__ float red[256];
    float s = 0.f;
    for (int c = threadIdx.x; c < DMODEL; c += 256) {
        float v = x[(size_t)row * DMODEL + c];
        s += v * v;
    }
    red[threadIdx.x] = s;
    __syncthreads();
    for (int off = 128; off > 0; off >>= 1) {
        if (threadIdx.x < off) red[threadIdx.x] += red[threadIdx.x + off];
        __syncthreads();
    }
    const float scale = rsqrtf(red[0] / (float)DMODEL + 1e-5f);
    for (int c = threadIdx.x; c < DMODEL; c += 256)
        xn[(size_t)row * DMODEL + c] = x[(size_t)row * DMODEL + c] * scale * nw[c];
}

// =====================================================================
// Causal depthwise conv (K=4) over time on xz[:, :DINNER], + SiLU -> xs
// =====================================================================
__global__ __launch_bounds__(256) void dwconv_silu(
    const float* __restrict__ xz, const float* __restrict__ cw,
    const float* __restrict__ cb, float* __restrict__ xs)
{
    const int idx = blockIdx.x * 256 + threadIdx.x;  // MROWS*DINNER
    const int d   = idx & (DINNER - 1);
    const int row = idx >> 10;                       // b*L + l
    const int l   = row & (LD - 1);
    const int bb  = row >> 9;
    float acc = cb[d];
    #pragma unroll
    for (int k = 0; k < 4; ++k) {
        const int ls = l - 3 + k;
        if (ls >= 0)
            acc += cw[d * 4 + k] * xz[((size_t)(bb * LD + ls)) * (2 * DINNER) + d];
    }
    xs[idx] = silu(acc);
}

// =====================================================================
// Selective scan + gating:
//   h[n] = exp(delta*A[d,n])*h[n] + delta*B[t,n]*xs ; y = sum_n h[n]*C[t,n]
//   out  = (y + xs*Dskip) * silu(z)
// 16 lanes per (b,d) channel (lane = state n), 16 channels per block.
// =====================================================================
__global__ __launch_bounds__(256) void mamba_scan(
    const float* __restrict__ delta, const float* __restrict__ xs,
    const float* __restrict__ xdb,   const float* __restrict__ xz,
    const float* __restrict__ alog,  const float* __restrict__ dskip,
    float* __restrict__ yg)
{
    const int tid = threadIdx.x;
    const int ch  = blockIdx.x * 16 + (tid >> 4);   // 0..4095
    const int n   = tid & 15;
    const int bb  = ch >> 10;
    const int d   = ch & (DINNER - 1);
    const float Acoef = -__expf(alog[d * STATE + n]);
    const float dsk   = dskip[d];
    float h = 0.f;
    for (int t = 0; t < LD; ++t) {
        const int row = bb * LD + t;
        const float dlt = delta[(size_t)row * DINNER + d];
        const float xsv = xs[(size_t)row * DINNER + d];
        const float Bm  = xdb[row * 64 + DTRANK + n];
        const float Cm  = xdb[row * 64 + DTRANK + STATE + n];
        const float dA  = __expf(dlt * Acoef);
        h = dA * h + dlt * Bm * xsv;
        float p = h * Cm;
        p += __shfl_xor(p, 1, 32);
        p += __shfl_xor(p, 2, 32);
        p += __shfl_xor(p, 4, 32);
        p += __shfl_xor(p, 8, 32);
        if (n == 0) {
            const float zv = xz[(size_t)row * (2 * DINNER) + DINNER + d];
            yg[(size_t)row * DINNER + d] = (p + xsv * dsk) * silu(zv);
        }
    }
}

// =====================================================================
// out_lens = max(feat_lens // 4, 1)
// =====================================================================
__global__ void lens_k(const int* __restrict__ feat_lens, int* __restrict__ out_lens)
{
    const int b = threadIdx.x;
    if (b < BD) {
        int v = feat_lens[b] / 4;
        out_lens[b] = v < 1 ? 1 : v;
    }
}

// =====================================================================
// Host orchestration
// =====================================================================
extern "C" void kernel_launch(void* const* d_in, const int* in_sizes, int n_in,
                              void* d_out, int out_size, void* d_ws, size_t ws_size,
                              hipStream_t stream)
{
    const float* feats     = (const float*)d_in[0];
    const int*   feat_lens = (const int*)  d_in[1];
    const float* conv1_w   = (const float*)d_in[2];
    const float* conv1_b   = (const float*)d_in[3];
    const float* conv2_w   = (const float*)d_in[4];
    const float* conv2_b   = (const float*)d_in[5];
    const float* norm_w    = (const float*)d_in[6];
    const float* in_proj_w = (const float*)d_in[7];
    const float* dwconv_w  = (const float*)d_in[8];
    const float* dwconv_b  = (const float*)d_in[9];
    const float* x_proj_w  = (const float*)d_in[10];
    const float* dt_proj_w = (const float*)d_in[11];
    const float* dt_proj_b = (const float*)d_in[12];
    const float* A_log     = (const float*)d_in[13];
    const float* Dskip     = (const float*)d_in[14];
    const float* out_proj_w= (const float*)d_in[15];
    const float* head_w    = (const float*)d_in[16];
    const float* head_b    = (const float*)d_in[17];
    (void)in_sizes; (void)n_in; (void)out_size; (void)ws_size;

    // ---------------- workspace layout ----------------
    float* ws    = (float*)d_ws;
    float* y1    = ws;                                 // 2,097,152 f32
    float* x     = y1    + (size_t)BD*512*1024;        // 1,048,576
    float* xn    = x     + (size_t)MROWS*DMODEL;       // 1,048,576
    float* xz    = xn    + (size_t)MROWS*DMODEL;       // 4,194,304
    float* xs    = xz    + (size_t)MROWS*2*DINNER;     // 2,097,152
    float* xdb   = xs    + (size_t)MROWS*DINNER;       //   131,072
    float* delta = xdb   + (size_t)MROWS*64;           // 2,097,152
    float* yg    = delta + (size_t)MROWS*DINNER;       // 2,097,152
    // f16 transposed weights after the f32 scratch
    _Float16* ipw16 = (_Float16*)(yg + (size_t)MROWS*DINNER);
    _Float16* xpw16 = ipw16 + (size_t)NBLK * DMODEL * 2 * DINNER;   // 8,388,608 h
    _Float16* dpw16 = xpw16 + (size_t)NBLK * DINNER * 64;           //   524,288 h
    _Float16* opw16 = dpw16 + (size_t)NBLK * DTRANK * DINNER;       //   262,144 h
    _Float16* hw16  = opw16 + (size_t)NBLK * DINNER * DMODEL;       // 4,194,304 h

    // ---------------- weight transpose/convert (per launch) ----------------
    dim3 tb(32, 8);
    transpose_cvt_f16<<<dim3(2*DINNER/32, DMODEL/32, NBLK), tb, 0, stream>>>(
        in_proj_w, ipw16, DMODEL, 2*DINNER);
    transpose_cvt_f16<<<dim3(64/32, DINNER/32, NBLK), tb, 0, stream>>>(
        x_proj_w, xpw16, DINNER, 64);
    transpose_cvt_f16<<<dim3(DINNER/32, DTRANK/32, NBLK), tb, 0, stream>>>(
        dt_proj_w, dpw16, DTRANK, DINNER);
    transpose_cvt_f16<<<dim3(DMODEL/32, DINNER/32, NBLK), tb, 0, stream>>>(
        out_proj_w, opw16, DINNER, DMODEL);
    transpose_cvt_f16<<<dim3(VOCAB/32, DMODEL/32, 1), tb, 0, stream>>>(
        head_w, hw16, DMODEL, VOCAB);

    // ---------------- frontend ----------------
    frontend_conv1<<<(BD*512*1024)/256, 256, 0, stream>>>(feats, conv1_w, conv1_b, y1);
    frontend_conv2<<<(BD*512*512)/256, 256, 0, stream>>>(y1, conv2_w, conv2_b, x);

    // ---------------- mamba blocks ----------------
    for (int blk = 0; blk < NBLK; ++blk) {
        rmsnorm_k<<<MROWS, 256, 0, stream>>>(x, norm_w + (size_t)blk * DMODEL, xn);

        // in_proj: (2048,512) x (512,2048) -> xz
        gemm_wmma_f16<true><<<dim3(2*DINNER/128, MROWS/64), 256, 0, stream>>>(
            xn, DMODEL,
            ipw16 + (size_t)blk * DMODEL * 2 * DINNER, DMODEL,
            xz, 2 * DINNER, MROWS, 2 * DINNER, DMODEL, nullptr, 0, 0);

        // causal depthwise conv + SiLU -> xs
        dwconv_silu<<<(MROWS*DINNER)/256, 256, 0, stream>>>(
            xz, dwconv_w + (size_t)blk * DINNER * 4, dwconv_b + (size_t)blk * DINNER, xs);

        // x_proj: (2048,1024) x (1024,64) -> xdb   (N=64 < 128 -> guarded)
        gemm_wmma_f16<false><<<dim3(1, MROWS/64), 256, 0, stream>>>(
            xs, DINNER,
            xpw16 + (size_t)blk * DINNER * 64, DINNER,
            xdb, 64, MROWS, 64, DINNER, nullptr, 0, 0);

        // dt_proj + bias + softplus: (2048,32) x (32,1024) -> delta
        gemm_wmma_f16<true><<<dim3(DINNER/128, MROWS/64), 256, 0, stream>>>(
            xdb, 64,
            dpw16 + (size_t)blk * DTRANK * DINNER, DTRANK,
            delta, DINNER, MROWS, DINNER, DTRANK,
            dt_proj_b + (size_t)blk * DINNER, 1, 0);

        // selective scan + gating -> yg
        mamba_scan<<<(BD*DINNER)/16, 256, 0, stream>>>(
            delta, xs, xdb, xz,
            A_log + (size_t)blk * DINNER * STATE,
            Dskip + (size_t)blk * DINNER, yg);

        // out_proj (+residual accumulate into x): (2048,1024) x (1024,512)
        gemm_wmma_f16<true><<<dim3(DMODEL/128, MROWS/64), 256, 0, stream>>>(
            yg, DINNER,
            opw16 + (size_t)blk * DINNER * DMODEL, DINNER,
            x, DMODEL, MROWS, DMODEL, DINNER, nullptr, 0, 1);
    }

    // ---------------- CTC head ----------------
    float* logits = (float*)d_out;
    gemm_wmma_f16<true><<<dim3(VOCAB/128, MROWS/64), 256, 0, stream>>>(
        x, DMODEL, hw16, DMODEL, logits, VOCAB,
        MROWS, VOCAB, DMODEL, head_b, 0, 0);

    // out_lens (int32 bits appended after logits)
    lens_k<<<1, 32, 0, stream>>>(feat_lens, (int*)d_out + (size_t)MROWS * VOCAB);
}
